// EfficientHebbian_26542897889392
// MI455X (gfx1250) — matrix operand order repaired and checked by reading, not compiled, not run on hardware
//
#include <hip/hip_runtime.h>
#include <hip/hip_bf16.h>
#include <stdint.h>

// ---------------------------------------------------------------------------
// Shapes (fixed by the reference)
// ---------------------------------------------------------------------------
#define DIN   1024
#define DOUT  1024
#define NTOK  16384          // B*T = 8*2048
#define HEBB  0.03125f       // 1/sqrt(1024)

typedef __attribute__((ext_vector_type(16))) __bf16 v16bf;
typedef __attribute__((ext_vector_type(8)))  float  v8f;
typedef __attribute__((ext_vector_type(2)))  float  v2f;

// Exact pointee type expected by __builtin_amdgcn_global_load_async_to_lds_b128:
// "__attribute__((__vector_size__(4 * sizeof(int)))) int" in AS1 / AS3.
typedef int v4i_vs __attribute__((vector_size(4 * sizeof(int))));
typedef __attribute__((address_space(1))) v4i_vs gvec_t;
typedef __attribute__((address_space(3))) v4i_vs lvec_t;
typedef __attribute__((address_space(3))) void  lds_void;

union FragBF { uint4 u[2]; v16bf v; };

static __device__ __forceinline__ unsigned short f2bf(float f) {
    unsigned u = __float_as_uint(f);
    u = (u + 0x7FFFu + ((u >> 16) & 1u)) >> 16;   // round-to-nearest-even
    return (unsigned short)u;
}

// ---------------------------------------------------------------------------
// CDNA5 async global->LDS (ASYNCcnt) helpers
// ---------------------------------------------------------------------------
static __device__ __forceinline__ void async_ld_b128(const void* g, void* l) {
#if __has_builtin(__builtin_amdgcn_global_load_async_to_lds_b128)
    __builtin_amdgcn_global_load_async_to_lds_b128(
        (gvec_t*)g, (lvec_t*)l, 0, 0);
#else
    unsigned loff = (unsigned)(unsigned long long)(lds_void*)l;
    asm volatile("global_load_async_to_lds_b128 %0, %1, off"
                 :: "v"(loff), "v"(g) : "memory");
#endif
}

static __device__ __forceinline__ void wait_async0() {
#if __has_builtin(__builtin_amdgcn_s_wait_asynccnt)
    __builtin_amdgcn_s_wait_asynccnt(0);
#else
    asm volatile("s_wait_asynccnt 0x0" ::: "memory");
#endif
}

static __device__ __forceinline__ void wait_async4() {
#if __has_builtin(__builtin_amdgcn_s_wait_asynccnt)
    __builtin_amdgcn_s_wait_asynccnt(4);
#else
    asm volatile("s_wait_asynccnt 0x4" ::: "memory");
#endif
}

// ---------------------------------------------------------------------------
// 1. LayerNorm: one block per token row, bf16 output (row-major)
// ---------------------------------------------------------------------------
__global__ void ln_kernel(const float* __restrict__ x,
                          const float* __restrict__ gamma,
                          const float* __restrict__ beta,
                          unsigned short* __restrict__ xn) {
    __shared__ float s1[256], s2[256];
    const int row = blockIdx.x, tid = threadIdx.x;
    const float4 v = *(const float4*)(x + (size_t)row * DIN + tid * 4);
    s1[tid] = v.x + v.y + v.z + v.w;
    s2[tid] = v.x * v.x + v.y * v.y + v.z * v.z + v.w * v.w;
    __syncthreads();
    for (int o = 128; o > 0; o >>= 1) {
        if (tid < o) { s1[tid] += s1[tid + o]; s2[tid] += s2[tid + o]; }
        __syncthreads();
    }
    const float mu  = s1[0] * (1.0f / DIN);
    const float var = s2[0] * (1.0f / DIN) - mu * mu;
    const float rs  = rsqrtf(var + 1e-5f);
    const int c = tid * 4;
    float o0 = (v.x - mu) * rs * gamma[c + 0] + beta[c + 0];
    float o1 = (v.y - mu) * rs * gamma[c + 1] + beta[c + 1];
    float o2 = (v.z - mu) * rs * gamma[c + 2] + beta[c + 2];
    float o3 = (v.w - mu) * rs * gamma[c + 3] + beta[c + 3];
    uint2 pk;
    pk.x = (unsigned)f2bf(o0) | ((unsigned)f2bf(o1) << 16);
    pk.y = (unsigned)f2bf(o2) | ((unsigned)f2bf(o3) << 16);
    *(uint2*)(xn + (size_t)row * DIN + c) = pk;
}

// ---------------------------------------------------------------------------
// 2. Transposes (LDS tiled, 32x32)
// ---------------------------------------------------------------------------
__global__ void transpose_bf16(const unsigned short* __restrict__ in,
                               unsigned short* __restrict__ out, int R, int C) {
    __shared__ unsigned short tile[32][33];
    const int tx = threadIdx.x, ty = threadIdx.y;
    const int c0 = blockIdx.x * 32, r0 = blockIdx.y * 32;
    #pragma unroll
    for (int j = 0; j < 4; ++j)
        tile[ty + j * 8][tx] = in[(size_t)(r0 + ty + j * 8) * C + c0 + tx];
    __syncthreads();
    #pragma unroll
    for (int j = 0; j < 4; ++j)
        out[(size_t)(c0 + ty + j * 8) * R + r0 + tx] = tile[tx][ty + j * 8];
}

__global__ void transpose_f32(const float* __restrict__ in,
                              float* __restrict__ out, int R, int C) {
    __shared__ float tile[32][33];
    const int tx = threadIdx.x, ty = threadIdx.y;
    const int c0 = blockIdx.x * 32, r0 = blockIdx.y * 32;
    #pragma unroll
    for (int j = 0; j < 4; ++j)
        tile[ty + j * 8][tx] = in[(size_t)(r0 + ty + j * 8) * C + c0 + tx];
    __syncthreads();
    #pragma unroll
    for (int j = 0; j < 4; ++j)
        out[(size_t)(c0 + ty + j * 8) * R + r0 + tx] = tile[tx][ty + j * 8];
}

__global__ void f32_to_bf16_kernel(const float* __restrict__ in,
                                   unsigned short* __restrict__ out, int n) {
    const int i = blockIdx.x * 256 + threadIdx.x;
    if (i < n) out[i] = f2bf(in[i]);
}

// ---------------------------------------------------------------------------
// 3. bf16 NT GEMM with async-LDS double-buffered pipeline.
//    C[m,n] = sum_k A[m,k] * Bt[n,k], f32 accumulate.
//    Block tile 128x128, 8 waves; wave tile 64x32 = 4x2 WMMA accumulators.
//    K staged 32 at a time via GLOBAL_LOAD_ASYNC_TO_LDS_B128 (ASYNCcnt),
//    double buffered: stage s+1 DMA overlaps stage s WMMAs.
//    MODE 0: y f32 -> yout[m,n]; also yT bf16 (transposed) for the trace GEMM.
//    MODE 1: fused Hebbian epilogue: new_trace, clip, weight pre-update.
// ---------------------------------------------------------------------------
#define LDSPITCH 40   // halves per row (32 data + 8 pad), 16B aligned

static __device__ __forceinline__ void stage_tile(
        const unsigned short* __restrict__ A,
        const unsigned short* __restrict__ Bt,
        int K, int m0, int n0, int kk, int tid,
        unsigned short (*la)[LDSPITCH], unsigned short (*lb)[LDSPITCH]) {
    #pragma unroll
    for (int t = 0; t < 2; ++t) {
        const int e = tid + t * 256;          // 0..511 -> 128 rows x 4 chunks
        const int row = e >> 2, ch = (e & 3) * 8;
        async_ld_b128(A + (size_t)(m0 + row) * K + kk + ch, &la[row][ch]);
    }
    #pragma unroll
    for (int t = 0; t < 2; ++t) {
        const int e = tid + t * 256;
        const int row = e >> 2, ch = (e & 3) * 8;
        async_ld_b128(Bt + (size_t)(n0 + row) * K + kk + ch, &lb[row][ch]);
    }
}

template <int MODE>
__global__ void gemm_bf16_nt(const unsigned short* __restrict__ A,
                             const unsigned short* __restrict__ Bt,
                             int K, int Ncols,
                             float* __restrict__ yout,
                             unsigned short* __restrict__ yT, int Mrows,
                             const float* __restrict__ trace,
                             float* __restrict__ otrace,
                             const float* __restrict__ weight,
                             float* __restrict__ wpre) {
    __shared__ unsigned short ldsA[2][128][LDSPITCH];   // 20 KB
    __shared__ unsigned short ldsB[2][128][LDSPITCH];   // 20 KB

    const int tid  = threadIdx.x;
    const int lane = tid & 31;
    const int wave = tid >> 5;
    const int wr = wave >> 2;          // 0..1  -> 64-row slab
    const int wc = wave & 3;           // 0..3  -> 32-col slab
    const int n0 = blockIdx.x * 128;
    const int m0 = blockIdx.y * 128;
    const int hi = lane >> 4, lo = lane & 15;
    const int ka = hi * 8;     // A frag: halves 0-7 = K ka..ka+7, 8-15 = ka+16..+23
    const int kb = hi * 8;     // B frag: halves 0-7 = K kb..kb+7, 8-15 = kb+16..+23

    v8f acc[4][2];
    #pragma unroll
    for (int mt = 0; mt < 4; ++mt)
        #pragma unroll
        for (int nt = 0; nt < 2; ++nt)
            acc[mt][nt] = (v8f){};

    const int nk = K / 32;
    stage_tile(A, Bt, K, m0, n0, 0, tid, ldsA[0], ldsB[0]);

    for (int s = 0; s < nk; ++s) {
        const int p = s & 1;
        const bool more = (s + 1 < nk);
        if (more)
            stage_tile(A, Bt, K, m0, n0, (s + 1) * 32, tid,
                       ldsA[p ^ 1], ldsB[p ^ 1]);
        if (more) wait_async4(); else wait_async0();   // stage s resident
        __syncthreads();

        FragBF a[4], b[2];
        #pragma unroll
        for (int mt = 0; mt < 4; ++mt) {
            const int r = wr * 64 + mt * 16 + lo;
            a[mt].u[0] = *(const uint4*)&ldsA[p][r][ka];
            a[mt].u[1] = *(const uint4*)&ldsA[p][r][ka + 16];
        }
        #pragma unroll
        for (int nt = 0; nt < 2; ++nt) {
            const int r = wc * 32 + nt * 16 + lo;
            b[nt].u[0] = *(const uint4*)&ldsB[p][r][kb];
            b[nt].u[1] = *(const uint4*)&ldsB[p][r][kb + 16];
        }
        #pragma unroll
        for (int mt = 0; mt < 4; ++mt)
            #pragma unroll
            for (int nt = 0; nt < 2; ++nt)
                acc[mt][nt] = __builtin_amdgcn_wmma_f32_16x16x32_bf16(
                    false, a[mt].v, false, b[nt].v, (short)0,
                    acc[mt][nt], false, false);

        __syncthreads();   // protect buffer p before stage s+2 overwrites it
    }

    // ---------------- epilogue ----------------
    #pragma unroll
    for (int mt = 0; mt < 4; ++mt) {
        #pragma unroll
        for (int nt = 0; nt < 2; ++nt) {
            const int m0w = m0 + wr * 64 + mt * 16;
            const int col = n0 + wc * 32 + nt * 16 + lo;
            if (MODE == 0) {
                unsigned short h[8];
                #pragma unroll
                for (int r = 0; r < 8; ++r) {
                    const int m = m0w + hi * 8 + r;
                    yout[(size_t)m * Ncols + col] = acc[mt][nt][r];
                    h[r] = f2bf(acc[mt][nt][r]);
                }
                uint4 pk;
                pk.x = (unsigned)h[0] | ((unsigned)h[1] << 16);
                pk.y = (unsigned)h[2] | ((unsigned)h[3] << 16);
                pk.z = (unsigned)h[4] | ((unsigned)h[5] << 16);
                pk.w = (unsigned)h[6] | ((unsigned)h[7] << 16);
                *(uint4*)(yT + (size_t)col * Mrows + m0w + hi * 8) = pk;
            } else {
                #pragma unroll
                for (int r = 0; r < 8; ++r) {
                    const int m = m0w + hi * 8 + r;
                    const size_t idx = (size_t)m * Ncols + col;
                    const float nt2 = 0.9f * trace[idx] + HEBB * acc[mt][nt][r];
                    otrace[idx] = nt2;
                    const float u = fminf(fmaxf(nt2, -1.0f), 1.0f);
                    wpre[idx] = weight[idx] + 0.01f * u;
                }
            }
        }
    }
}

// ---------------------------------------------------------------------------
// 4. f32 NT GEMM (Newton-Schulz): v_wmma_f32_16x16x4_f32.
//    MODE 0: C = A*Bt^T          (H = Z Z^T with Bt = Z)
//    MODE 1: C = 1.5*Zold - 0.5*A*Bt^T   (polar iteration step)
// ---------------------------------------------------------------------------
template <int MODE>
__global__ void gemm_f32_nt(const float* __restrict__ A,
                            const float* __restrict__ Bt,
                            float* __restrict__ C,
                            const float* __restrict__ Zold, int K, int N) {
    const int lane = threadIdx.x & 31;
    const int wave = threadIdx.x >> 5;
    const int n0 = blockIdx.x * 128 + wave * 16;
    const int m0 = blockIdx.y * 16;
    const int hi = lane >> 4, lo = lane & 15;
    const size_t arow = (size_t)(m0 + lo) * K;
    const size_t brow = (size_t)(n0 + lo) * K;
    const int kk = hi * 2;   // f32 A/B: VGPR0/1 = K kk, kk+1
    v8f acc = {};
    #pragma unroll 8
    for (int k = 0; k < K; k += 4) {
        if ((k & 63) == 0 && k + 256 < K) {
            __builtin_prefetch(A  + arow + k + 256, 0, 0);  // global_prefetch_b8
            __builtin_prefetch(Bt + brow + k + 256, 0, 0);
        }
        const v2f a = *(const v2f*)(A  + arow + k + kk);
        const v2f b = *(const v2f*)(Bt + brow + k + kk);
        acc = __builtin_amdgcn_wmma_f32_16x16x4_f32(
                  false, a, false, b, (short)0, acc, false, false);
    }
    const int col = n0 + lo;
    #pragma unroll
    for (int r = 0; r < 8; ++r) {
        const int m = m0 + hi * 8 + r;
        const size_t idx = (size_t)m * N + col;
        if (MODE == 0) C[idx] = acc[r];
        else           C[idx] = 1.5f * Zold[idx] - 0.5f * acc[r];
    }
}

// ---------------------------------------------------------------------------
// 5. Column norms (+ Frobenius accumulator) and global scaling
// ---------------------------------------------------------------------------
__global__ void zero_kernel(float* p) { *p = 0.0f; }

__global__ void colnorm_kernel(const float* __restrict__ wpre,
                               float* __restrict__ z,
                               float* __restrict__ fro2) {
    __shared__ float red[256];
    __shared__ float denom[64];
    const int c  = blockIdx.x * 64 + (threadIdx.x & 63);
    const int r0 = threadIdx.x >> 6;
    float ss = 0.0f;
    for (int r = r0; r < DOUT; r += 4) {
        const float v = wpre[(size_t)r * DIN + c];
        ss += v * v;
    }
    red[threadIdx.x] = ss;
    __syncthreads();
    if (threadIdx.x < 64) {
        const float t = red[threadIdx.x] + red[threadIdx.x + 64] +
                        red[threadIdx.x + 128] + red[threadIdx.x + 192];
        const float nrm = sqrtf(t);
        denom[threadIdx.x] = fmaxf(nrm, 1e-3f);
        const float contrib = (nrm >= 1e-3f) ? 1.0f : (t * 1e6f);
        atomicAdd(fro2, contrib);
    }
    __syncthreads();
    const float d = denom[threadIdx.x & 63];
    for (int r = r0; r < DOUT; r += 4) {
        const size_t idx = (size_t)r * DIN + c;
        z[idx] = wpre[idx] / d;
    }
}

__global__ void scale_kernel(float* __restrict__ z, const float* __restrict__ fro2) {
    const int i = blockIdx.x * 256 + threadIdx.x;
    const float s = rsqrtf(fmaxf(*fro2, 1e-12f));
    z[i] *= s;
}

// ---------------------------------------------------------------------------
// Host orchestration
// ---------------------------------------------------------------------------
extern "C" void kernel_launch(void* const* d_in, const int* in_sizes, int n_in,
                              void* d_out, int out_size, void* d_ws, size_t ws_size,
                              hipStream_t stream) {
    (void)in_sizes; (void)n_in; (void)out_size; (void)ws_size;
    const float* x      = (const float*)d_in[0];
    const float* weight = (const float*)d_in[1];
    const float* gamma  = (const float*)d_in[2];
    const float* beta   = (const float*)d_in[3];
    const float* trace  = (const float*)d_in[4];

    float* y          = (float*)d_out;                 // [16384,1024]
    float* new_weight = y + (size_t)NTOK * DOUT;       // [1024,1024]
    float* new_trace  = new_weight + (size_t)DOUT * DIN;

    char* ws = (char*)d_ws;
    size_t off = 0;
    unsigned short* xn  = (unsigned short*)(ws + off); off += (size_t)NTOK * DIN * 2;
    unsigned short* xnT = (unsigned short*)(ws + off); off += (size_t)DIN * NTOK * 2;
    unsigned short* yT  = (unsigned short*)(ws + off); off += (size_t)DOUT * NTOK * 2;
    unsigned short* wbf = (unsigned short*)(ws + off); off += (size_t)DOUT * DIN * 2;
    float* wpre = (float*)(ws + off); off += (size_t)DOUT * DIN * 4;
    float* za   = (float*)(ws + off); off += (size_t)DOUT * DIN * 4;
    float* zb   = (float*)(ws + off); off += (size_t)DOUT * DIN * 4;
    float* zt   = (float*)(ws + off); off += (size_t)DOUT * DIN * 4;
    float* hh   = (float*)(ws + off); off += (size_t)DOUT * DOUT * 4;
    float* fro2 = (float*)(ws + off); off += 256;

    // 1) LayerNorm -> bf16
    ln_kernel<<<NTOK, 256, 0, stream>>>(x, gamma, beta, xn);
    // 2) xn^T for trace GEMM; weight -> bf16 (Bt operand of GEMM1)
    transpose_bf16<<<dim3(DIN / 32, NTOK / 32), dim3(32, 8), 0, stream>>>(xn, xnT, NTOK, DIN);
    f32_to_bf16_kernel<<<(DOUT * DIN) / 256, 256, 0, stream>>>(weight, wbf, DOUT * DIN);
    // 3) y = xn * W^T  (NT: Bt = W row-major), emits y f32 + yT bf16
    gemm_bf16_nt<0><<<dim3(DOUT / 128, NTOK / 128), 256, 0, stream>>>(
        xn, wbf, DIN, DOUT, y, yT, NTOK,
        nullptr, nullptr, nullptr, nullptr);
    // 4) new_trace = 0.9*trace + hebb*(y^T xn); fused clip + weight pre-update
    gemm_bf16_nt<1><<<dim3(DIN / 128, DOUT / 128), 256, 0, stream>>>(
        yT, xnT, NTOK, DIN, nullptr, nullptr, 0,
        trace, new_trace, weight, wpre);
    // 5) column normalize + Frobenius pre-scale for Newton-Schulz
    zero_kernel<<<1, 1, 0, stream>>>(fro2);
    colnorm_kernel<<<DIN / 64, 256, 0, stream>>>(wpre, za, fro2);
    scale_kernel<<<(DOUT * DIN) / 256, 256, 0, stream>>>(za, fro2);
    // 6) Polar factor (U V^T) via 5 Newton-Schulz iterations: Z <- 1.5Z - 0.5(ZZ^T)Z
    float* cur = za; float* nxt = zb;
    for (int it = 0; it < 5; ++it) {
        transpose_f32<<<dim3(32, 32), dim3(32, 8), 0, stream>>>(cur, zt, DOUT, DIN);
        gemm_f32_nt<0><<<dim3(DOUT / 128, DOUT / 16), 256, 0, stream>>>(
            cur, cur, hh, nullptr, DIN, DOUT);               // H = Z Z^T
        float* dst = (it == 4) ? new_weight : nxt;
        gemm_f32_nt<1><<<dim3(DIN / 128, DOUT / 16), 256, 0, stream>>>(
            hh, zt, dst, cur, DOUT, DIN);                    // 1.5Z - 0.5 H Z
        float* t = cur; cur = nxt; nxt = t;
    }
}
